// VQEmbeddingEMA_30760555774510
// MI455X (gfx1250) — matrix operand order repaired and checked by reading, not compiled, not run on hardware
//
#include <hip/hip_runtime.h>
#include <hip/hip_bf16.h>
#include <math.h>

// Problem sizes (fixed by the reference)
#define N_ROWS   32768   // 16 * 2048
#define M_CODES  4096
#define K_DIM    256
#define Q_ELEMS  (N_ROWS * K_DIM)   // 8388608
#define BPAD     264                // 256 + 8 bf16 pad -> conflict-free LDS reads

typedef __attribute__((ext_vector_type(16))) __bf16 v16bf;
typedef __attribute__((ext_vector_type(8)))  float  v8f;

union BF16Frag {
    v16bf v;
    unsigned short u[16];
};

// f32 -> bf16 round-to-nearest-even (bit manipulation; no scalar __bf16 math needed)
__device__ __forceinline__ unsigned short f2bf(float f) {
    unsigned u = __float_as_uint(f);
    unsigned r = u + 0x7FFFu + ((u >> 16) & 1u);
    return (unsigned short)(r >> 16);
}
__device__ __forceinline__ float bf2f(unsigned short h) {
    return __uint_as_float(((unsigned)h) << 16);
}

// Async global->LDS copy of 64 contiguous bytes (4 x B128), GVS addressing:
//   mem  = SGPR base + 32-bit VGPR offset + INST_OFFSET
//   lds  = LDS_BASE + VGPR lds addr      + INST_OFFSET   (offset added to both)
// Tracked by ASYNCcnt; completion enforced with s_wait_asynccnt.
__device__ __forceinline__ void async_copy64(unsigned lds_addr, unsigned goff,
                                             const float* base) {
    asm volatile(
        "global_load_async_to_lds_b128 %0, %1, %2 offset:0\n\t"
        "global_load_async_to_lds_b128 %0, %1, %2 offset:16\n\t"
        "global_load_async_to_lds_b128 %0, %1, %2 offset:32\n\t"
        "global_load_async_to_lds_b128 %0, %1, %2 offset:48"
        :
        : "v"(lds_addr), "v"(goff), "s"((unsigned long long)base)
        : "memory");
}

__device__ __forceinline__ void wait_async0() {
    asm volatile("s_wait_asynccnt 0x0" ::: "memory");
}

// ---------------------------------------------------------------------------
// Kernel 0: zero histogram + loss accumulators, compute ||e||^2 per code.
// grid = 16 x 256 (one thread per code)
// ---------------------------------------------------------------------------
__global__ __launch_bounds__(256)
void vq_init(const float* __restrict__ emb,
             float* __restrict__ counts,
             float* __restrict__ enorm,
             float* __restrict__ accum) {
    const int gid = blockIdx.x * 256 + threadIdx.x;   // 0..4095
    if (gid < 2) accum[gid] = 0.0f;
    counts[gid] = 0.0f;
    const float4* e = (const float4*)(emb + (size_t)gid * K_DIM);
    float s = 0.0f;
#pragma unroll 8
    for (int i = 0; i < K_DIM / 4; ++i) {
        float4 v = e[i];
        s += v.x * v.x + v.y * v.y + v.z * v.z + v.w * v.w;
    }
    enorm[gid] = s;
}

// ---------------------------------------------------------------------------
// Kernel 1: fused GEMM (split-bf16 WMMA) + per-row argmin over all 4096 codes.
//   - workgroup = 256 threads = 8 waves, each wave owns a 16-row tile
//   - workgroup tile = 128 rows; sweeps all 256 column chunks of 16 codes
//   - A fragments (hi+lo) resident in registers
//   - B panel DMA'd global->LDS with async-LDS loads, double buffered one
//     chunk ahead (ASYNCcnt), then converted LDS f32 -> LDS bf16 hi/lo
//   - two independent WMMA accumulator chains (hi*hi | cross terms)
//   dist = ||e||^2 - 2 * x.e   (||x||^2 dropped: row-constant for argmin)
// ---------------------------------------------------------------------------
__global__ __launch_bounds__(256, 1)
void vq_gemm_argmin(const float* __restrict__ x,
                    const float* __restrict__ emb,
                    const float* __restrict__ enorm,
                    float* __restrict__ idx_out) {
    __shared__ __align__(16) float          sF[2][16][K_DIM];   // raw f32 panels
    __shared__ __align__(16) unsigned short sBhi[16][BPAD];
    __shared__ __align__(16) unsigned short sBlo[16][BPAD];
    __shared__ float sEn[16];

    const int t      = threadIdx.x;
    const int lane   = t & 31;
    const int waveid = t >> 5;
    const int hlf    = lane >> 4;   // 0: lanes 0-15, 1: lanes 16-31
    const int mrow   = lane & 15;

    // staging role: thread t owns 16 contiguous k's (64 bytes) of one code
    const int cl = t >> 4;          // local code 0..15
    const int ks = (t & 15) * 16;   // k start

    // ---- prologue: kick off async DMA of panel 0 --------------------------
    async_copy64((unsigned)(size_t)&sF[0][cl][ks],
                 (unsigned)(((unsigned)cl * K_DIM + ks) * sizeof(float)), emb);

    // ---- Build resident A fragments: 16 rows x 256 k, bf16 hi + lo --------
    // ISA 16-bit A 16x32 layout: lanes 0-15 hold K {0-7,16-23}, lanes 16-31
    // hold K {8-15,24-31} (per 32-wide k-chunk), row M = lane & 15.
    const int rowbase = blockIdx.x * 128 + waveid * 16;
    const float* xrow = x + (size_t)(rowbase + mrow) * K_DIM;
    BF16Frag ah[8], al[8];
#pragma unroll
    for (int kc = 0; kc < 8; ++kc) {
        const int kb = kc * 32 + hlf * 8;
        float v[16];
        float4 p;
        p = *(const float4*)(xrow + kb);      v[0]=p.x; v[1]=p.y; v[2]=p.z; v[3]=p.w;
        p = *(const float4*)(xrow + kb + 4);  v[4]=p.x; v[5]=p.y; v[6]=p.z; v[7]=p.w;
        p = *(const float4*)(xrow + kb + 16); v[8]=p.x; v[9]=p.y; v[10]=p.z; v[11]=p.w;
        p = *(const float4*)(xrow + kb + 20); v[12]=p.x; v[13]=p.y; v[14]=p.z; v[15]=p.w;
#pragma unroll
        for (int e = 0; e < 16; ++e) {
            unsigned short h = f2bf(v[e]);
            ah[kc].u[e] = h;
            al[kc].u[e] = f2bf(v[e] - bf2f(h));
        }
    }

    float minv[8];
    int   mini[8];
#pragma unroll
    for (int j = 0; j < 8; ++j) { minv[j] = 3.4e38f; mini[j] = 0; }

    for (int chunk = 0; chunk < M_CODES / 16; ++chunk) {
        const int colbase = chunk * 16;

        // panel `chunk` has landed in sF[chunk&1] once every thread's async
        // portion completes and the workgroup synchronizes
        wait_async0();
        __syncthreads();

        // ---- issue async DMA for the NEXT panel (overlaps with the WMMAs);
        //      its target buffer was last read at chunk-1, before the barrier
        if (chunk + 1 < M_CODES / 16) {
            async_copy64((unsigned)(size_t)&sF[(chunk + 1) & 1][cl][ks],
                         (unsigned)((((unsigned)(chunk + 1) * 16 + cl) * K_DIM + ks)
                                    * sizeof(float)),
                         emb);
        }

        // ---- convert raw f32 panel -> bf16 hi/lo panels (LDS -> LDS) ------
        {
            const float* src = &sF[chunk & 1][cl][ks];
            float v[16];
            float4 p;
            p = ((const float4*)src)[0]; v[0]=p.x; v[1]=p.y; v[2]=p.z; v[3]=p.w;
            p = ((const float4*)src)[1]; v[4]=p.x; v[5]=p.y; v[6]=p.z; v[7]=p.w;
            p = ((const float4*)src)[2]; v[8]=p.x; v[9]=p.y; v[10]=p.z; v[11]=p.w;
            p = ((const float4*)src)[3]; v[12]=p.x; v[13]=p.y; v[14]=p.z; v[15]=p.w;
            unsigned* dh = (unsigned*)&sBhi[cl][ks];
            unsigned* dl = (unsigned*)&sBlo[cl][ks];
#pragma unroll
            for (int i = 0; i < 8; ++i) {
                unsigned short h0 = f2bf(v[2 * i]);
                unsigned short h1 = f2bf(v[2 * i + 1]);
                dh[i] = (unsigned)h0 | ((unsigned)h1 << 16);
                unsigned short l0 = f2bf(v[2 * i] - bf2f(h0));
                unsigned short l1 = f2bf(v[2 * i + 1] - bf2f(h1));
                dl[i] = (unsigned)l0 | ((unsigned)l1 << 16);
            }
            if (t < 16) sEn[t] = enorm[colbase + t];
        }
        __syncthreads();

        // ---- K loop: 8 chunks of 32, 3 WMMAs each (hi*hi + hi*lo + lo*hi),
        //      split into two independent accumulator chains
        v8f acc0 = {0.0f, 0.0f, 0.0f, 0.0f, 0.0f, 0.0f, 0.0f, 0.0f};
        v8f acc1 = {0.0f, 0.0f, 0.0f, 0.0f, 0.0f, 0.0f, 0.0f, 0.0f};
#pragma unroll
        for (int kc = 0; kc < 8; ++kc) {
            // B 32x16 layout (matches A): lane N = lane&15; lanes 0-15 hold
            // K {0-7,16-23}, lanes 16-31 hold K {8-15,24-31} of the chunk.
            const int kbB = kc * 32 + hlf * 8;
            BF16Frag bh, bl;
            *(uint4*)&bh.u[0] = *(const uint4*)&sBhi[mrow][kbB];
            *(uint4*)&bh.u[8] = *(const uint4*)&sBhi[mrow][kbB + 16];
            *(uint4*)&bl.u[0] = *(const uint4*)&sBlo[mrow][kbB];
            *(uint4*)&bl.u[8] = *(const uint4*)&sBlo[mrow][kbB + 16];
            acc0 = __builtin_amdgcn_wmma_f32_16x16x32_bf16(
                       false, ah[kc].v, false, bh.v, (short)0, acc0, false, false);
            acc1 = __builtin_amdgcn_wmma_f32_16x16x32_bf16(
                       false, ah[kc].v, false, bl.v, (short)0, acc1, false, false);
            acc1 = __builtin_amdgcn_wmma_f32_16x16x32_bf16(
                       false, al[kc].v, false, bh.v, (short)0, acc1, false, false);
        }

        // ---- fused argmin update: C/D layout lane holds col N = lane&15,
        //      VGPR j holds row j (lanes 0-15) / j+8 (lanes 16-31).
        const float en   = sEn[mrow];
        const int   code = colbase + mrow;
#pragma unroll
        for (int j = 0; j < 8; ++j) {
            float dist = en - 2.0f * (acc0[j] + acc1[j]);
            if (dist < minv[j]) { minv[j] = dist; mini[j] = code; }
        }
    }

    // ---- cross-lane argmin within each 16-lane half (16 candidate cols) ---
#pragma unroll
    for (int j = 0; j < 8; ++j) {
        float v  = minv[j];
        int   ix = mini[j];
#pragma unroll
        for (int off = 8; off >= 1; off >>= 1) {
            float ov = __shfl_xor(v, off, 16);
            int   oi = __shfl_xor(ix, off, 16);
            if (ov < v || (ov == v && oi < ix)) { v = ov; ix = oi; }
        }
        if (mrow == 0) {
            const int r = rowbase + hlf * 8 + j;
            idx_out[r] = (float)ix;
        }
    }
}

// ---------------------------------------------------------------------------
// Kernel 2: gather quantized rows, per-row commitment-loss partials (block
// reduce), histogram. One 256-thread block per row. Bandwidth-bound (~64 MB).
// ---------------------------------------------------------------------------
__global__ __launch_bounds__(256)
void vq_gather_loss(const float* __restrict__ x,
                    const float* __restrict__ emb,
                    const float* __restrict__ idx_in,
                    float* __restrict__ qout,
                    float* __restrict__ counts,
                    float* __restrict__ accum) {
    const int row  = blockIdx.x;
    const int d    = threadIdx.x;
    const int code = (int)idx_in[row];
    const float q  = emb[(size_t)code * K_DIM + d];
    const float xv = x[(size_t)row * K_DIM + d];
    qout[(size_t)row * K_DIM + d] = q;   // quantized_st forward value == quantized

    float sq = (xv - q) * (xv - q);
    float ax = fabsf(xv);
#pragma unroll
    for (int off = 16; off >= 1; off >>= 1) {
        sq += __shfl_down(sq, off, 32);
        ax += __shfl_down(ax, off, 32);
    }
    __shared__ float r1[8], r2[8];
    const int lane = threadIdx.x & 31, w = threadIdx.x >> 5;
    if (lane == 0) { r1[w] = sq; r2[w] = ax; }
    __syncthreads();
    if (threadIdx.x == 0) {
        float s = 0.0f, a = 0.0f;
#pragma unroll
        for (int i = 0; i < 8; ++i) { s += r1[i]; a += r2[i]; }
        const float np = (a > 0.0f) ? 1.0f : 0.0f;
        atomicAdd(&accum[0], (s * (1.0f / (float)K_DIM)) * np);
        atomicAdd(&accum[1], np);
        atomicAdd(&counts[code], 1.0f);   // exact: integer-valued f32 adds
    }
}

// ---------------------------------------------------------------------------
// Kernel 3: scalars — loss and perplexity.
// ---------------------------------------------------------------------------
__global__ __launch_bounds__(256)
void vq_finalize(const float* __restrict__ counts,
                 const float* __restrict__ accum,
                 float* __restrict__ loss_out,
                 float* __restrict__ perp_out) {
    __shared__ float red[256];
    float s = 0.0f;
    for (int i = threadIdx.x; i < M_CODES; i += 256) {
        float p = counts[i] * (1.0f / (float)N_ROWS);
        s += p * logf(p + 1e-10f);
    }
    red[threadIdx.x] = s;
    __syncthreads();
    for (int off = 128; off >= 1; off >>= 1) {
        if (threadIdx.x < off) red[threadIdx.x] += red[threadIdx.x + off];
        __syncthreads();
    }
    if (threadIdx.x == 0) {
        *loss_out = 0.25f * accum[0] / accum[1];   // COMMITMENT_COST * e_latent_loss
        *perp_out = expf(-red[0]);
    }
}

// ---------------------------------------------------------------------------
// Host launcher
// ---------------------------------------------------------------------------
extern "C" void kernel_launch(void* const* d_in, const int* in_sizes, int n_in,
                              void* d_out, int out_size, void* d_ws, size_t ws_size,
                              hipStream_t stream) {
    (void)in_sizes; (void)n_in; (void)out_size; (void)ws_size;

    const float* x   = (const float*)d_in[0];   // [16,2048,256] f32
    const float* emb = (const float*)d_in[1];   // [4096,256]    f32

    // outputs flat in return order: quantized_st, loss, indices, perplexity
    float* out  = (float*)d_out;
    float* q    = out;                       // 8388608
    float* loss = out + (size_t)Q_ELEMS;     // 1
    float* idxo = loss + 1;                  // 32768 (indices as f32)
    float* perp = idxo + N_ROWS;             // 1

    // workspace: counts[4096] | enorm[4096] | accum[2]
    float* ws     = (float*)d_ws;
    float* counts = ws;
    float* enorm  = ws + M_CODES;
    float* accum  = ws + 2 * M_CODES;

    vq_init       <<<dim3(M_CODES / 256), dim3(256), 0, stream>>>(emb, counts, enorm, accum);
    vq_gemm_argmin<<<dim3(N_ROWS / 128),  dim3(256), 0, stream>>>(x, emb, enorm, idxo);
    vq_gather_loss<<<dim3(N_ROWS),        dim3(256), 0, stream>>>(x, emb, idxo, q, counts, accum);
    vq_finalize   <<<dim3(1),             dim3(256), 0, stream>>>(counts, accum, loss, perp);
}
